// EmbedMatcher_20340965114122
// MI455X (gfx1250) — matrix operand order repaired and checked by reading, not compiled, not run on hardware
//
#include <hip/hip_runtime.h>
#include <hip/hip_bf16.h>

// ---------------------------------------------------------------------------
// EmbedMatcher on MI455X (gfx1250, wave32, WMMA).
// All GEMMs use v_wmma_f32_16x16x32_f16 (f16 A/B, f32 acc).
// Algebraic simplifications (BS==1):
//   * softmax over a (B,1) axis is 1  ->  r == sg each step
//   * h_r @ Whh.T = hq @ WhhL.T + (sg @ WhhR.T)   (2nd term is a constant vec)
//   * qv@Wih.T + bih + bhh is step-invariant (gbase0); step 1 uses it directly
// Embedding table converted to f16 once (51 MB -> L2 resident) to halve
// gather bytes and feed WMMA directly.
// ---------------------------------------------------------------------------

typedef __attribute__((ext_vector_type(16))) _Float16 v16h;
typedef __attribute__((ext_vector_type(8)))  _Float16 v8h;
typedef __attribute__((ext_vector_type(8)))  float    v8f;

union V16U { v16h v; v8h h[2]; };

#define BQ   4096
#define NV   200000
#define DD   128
#define MNBR 50
#define DM2  256
#define HH   512
#define G4   2048

__device__ __forceinline__ float sigf(float x) { return 1.f / (1.f + __expf(-x)); }

// ------------------------------ converters ---------------------------------
__global__ void f32_to_f16_kernel(const float* __restrict__ src,
                                  _Float16* __restrict__ dst, long n) {
  long i = (long)blockIdx.x * blockDim.x + threadIdx.x;
  long st = (long)gridDim.x * blockDim.x;
  for (; i < n; i += st) dst[i] = (_Float16)src[i];
}

// dst(rows x cols, dense) = f16(src(rows x src_stride)[:, :cols])
__global__ void f32_to_f16_strided_kernel(const float* __restrict__ src,
                                          _Float16* __restrict__ dst,
                                          int rows, int cols, int src_stride) {
  long n = (long)rows * cols;
  long i = (long)blockIdx.x * blockDim.x + threadIdx.x;
  long st = (long)gridDim.x * blockDim.x;
  for (; i < n; i += st) {
    long r = i / cols; long c = i - r * cols;
    dst[i] = (_Float16)src[r * src_stride + c];
  }
}

// --------------------- fused nbr (gather + GEMM + attn) --------------------
// One block per batch row b: X(50x256 gathered f16) @ gcn_W^T -> proj(50x128),
// leakyrelu, attn softmax over M, gated mix with emb[self].
__global__ __launch_bounds__(256)
void nbr_wmma_kernel(const int* __restrict__ conn,      // (NB, 50, 2)
                     const int* __restrict__ self_ids,  // (NB, 2)
                     int self_col,
                     const _Float16* __restrict__ embh, // (V+1, 128) f16
                     const float* __restrict__ embf,    // (V+1, 128) f32
                     const _Float16* __restrict__ Wh,   // (128, 256) f16 [n][k]
                     const float* __restrict__ gcn_lb, const float* __restrict__ gcn_b,
                     const float* __restrict__ attnW, const float* __restrict__ attn_b,
                     const float* __restrict__ gateW, const float* __restrict__ gate_lb,
                     const float* __restrict__ gate_b,
                     float* __restrict__ out)           // (NB, 128)
{
  __shared__ __align__(16) _Float16 Xs[64][264];  // padded: 528B row stride
  __shared__ __align__(16) float    Ps[64][132];
  __shared__ float Sc[64], Wt[64], Agg[128], GateSh;

  const int b = blockIdx.x;
  const int t = threadIdx.x;

  // gather X = [emb[rel], emb[ent]] rows into LDS (rows 50..63 zero)
  {
    int row = t >> 2, part = t & 3;           // 4 threads/row, 64 halves each
    v8h* dst = (v8h*)&Xs[row][part * 64];
    if (row < MNBR) {
      int id = conn[((long)b * MNBR + row) * 2 + (part >> 1)];
      const v8h* src = (const v8h*)(embh + (long)id * DD + (part & 1) * 64);
#pragma unroll
      for (int i = 0; i < 8; ++i) dst[i] = src[i];
    } else {
      v8h z = {};
#pragma unroll
      for (int i = 0; i < 8; ++i) dst[i] = z;
    }
  }
  __syncthreads();

  const int wave = t >> 5, lane = t & 31;
  const int nlo = lane & 15;
  const bool hiHalf = (lane & 16) != 0;
  const int col = wave * 16 + nlo;            // output column 0..127

  // B fragments (register resident): 16 contiguous k per lane-half
  v16h bf[8];
  {
    const _Float16* base = Wh + (long)col * DM2 + (hiHalf ? 16 : 0);
#pragma unroll
    for (int kb = 0; kb < 8; ++kb) {
      const _Float16* p = base + kb * 32;
      V16U u; u.h[0] = *(const v8h*)p; u.h[1] = *(const v8h*)(p + 8);
      bf[kb] = u.v;
    }
  }

  const float lbb = gcn_lb[col] + gcn_b[col];
#pragma unroll
  for (int mt = 0; mt < 4; ++mt) {
    v8f acc = {};
#pragma unroll
    for (int kb = 0; kb < 8; ++kb) {
      // A fragment: 16 contiguous k per lane-half (matches B)
      const _Float16* p = &Xs[mt * 16 + nlo][kb * 32 + (hiHalf ? 16 : 0)];
      V16U u; u.h[0] = *(const v8h*)p; u.h[1] = *(const v8h*)(p + 8);
      acc = __builtin_amdgcn_wmma_f32_16x16x32_f16(false, u.v, false, bf[kb],
                                                   (short)0, acc, false, false);
    }
#pragma unroll
    for (int r = 0; r < 8; ++r) {             // D: M = r + 8*hiHalf, N = nlo
      int rowm = mt * 16 + r + (hiHalf ? 8 : 0);
      float v = acc[r] + lbb;
      Ps[rowm][col] = (v >= 0.f) ? v : 0.01f * v;   // leaky relu
    }
  }
  __syncthreads();

  // attention scores over M
  if (t < MNBR) {
    float s = attn_b[0];
    for (int n = 0; n < DD; ++n) s += Ps[t][n] * attnW[n];
    Sc[t] = s;
  }
  __syncthreads();
  if (t == 0) {                               // tiny softmax over 50
    float mx = Sc[0];
    for (int m = 1; m < MNBR; ++m) mx = fmaxf(mx, Sc[m]);
    float sum = 0.f;
    for (int m = 0; m < MNBR; ++m) { float e = __expf(Sc[m] - mx); Wt[m] = e; sum += e; }
    float inv = 1.f / sum;
    for (int m = 0; m < MNBR; ++m) Wt[m] *= inv;
  }
  __syncthreads();
  if (t < DD) {
    float a = 0.f;
    for (int m = 0; m < MNBR; ++m) a += Ps[m][t] * Wt[m];
    Agg[t] = a;
  }
  __syncthreads();
  if (t == 0) {
    float gi = gate_lb[0] + gate_b[0];
    for (int n = 0; n < DD; ++n) gi += Agg[n] * gateW[n];
    GateSh = sigf(gi);
  }
  __syncthreads();
  if (t < DD) {
    int sid = self_ids[(long)b * 2 + self_col];
    float ga = GateSh;
    out[(long)b * DD + t] = ga * Agg[t] + (1.f - ga) * embf[(long)sid * DD + t];
  }
}

// ---------------- qv = concat[(L1+L2)/2, (R1+R2)/2], also f16 --------------
__global__ void combine_pair_kernel(const float* __restrict__ L1, const float* __restrict__ L2,
                                    const float* __restrict__ R1, const float* __restrict__ R2,
                                    float* __restrict__ qv, _Float16* __restrict__ qvh, long nb) {
  long total = nb * DM2;
  long i = (long)blockIdx.x * blockDim.x + threadIdx.x;
  long st = (long)gridDim.x * blockDim.x;
  for (; i < total; i += st) {
    long b = i >> 8; int n = (int)(i & 255);
    float v = (n < DD) ? 0.5f * (L1[b * DD + n] + L2[b * DD + n])
                       : 0.5f * (R1[b * DD + n - DD] + R2[b * DD + n - DD]);
    qv[i] = v; qvh[i] = (_Float16)v;
  }
}

// ---------------- support MLP + LayerNorm -> sg (one block) ----------------
__global__ __launch_bounds__(512)
void support_mlp_kernel(const float* __restrict__ sv,
                        const float* __restrict__ p1W, const float* __restrict__ p1b,
                        const float* __restrict__ p2W, const float* __restrict__ p2b,
                        const float* __restrict__ ln_a, const float* __restrict__ ln_b,
                        float* __restrict__ sg) {
  __shared__ float h1[HH], h2[DM2], stats[2];
  int t = threadIdx.x;
  {
    float a = p1b[t];
    for (int k = 0; k < DM2; ++k) a += sv[k] * p1W[(long)t * DM2 + k];
    h1[t] = a > 0.f ? a : 0.f;
  }
  __syncthreads();
  if (t < DM2) {
    float a = p2b[t] + sv[t];
    for (int k = 0; k < HH; ++k) a += h1[k] * p2W[(long)t * HH + k];
    h2[t] = a;
  }
  __syncthreads();
  if (t == 0) {
    float mu = 0.f;
    for (int k = 0; k < DM2; ++k) mu += h2[k];
    mu /= (float)DM2;
    float var = 0.f;
    for (int k = 0; k < DM2; ++k) { float d = h2[k] - mu; var += d * d; }
    var /= (float)(DM2 - 1);                  // ddof=1
    stats[0] = mu; stats[1] = sqrtf(var);
  }
  __syncthreads();
  if (t < DM2) sg[t] = (h2[t] - stats[0]) / (stats[1] + 0.001f) * ln_a[t] + ln_b[t];
}

// ---- biasvec = bih+bhh ; svec = sg @ Whh[:,256:].T (constant per step) ----
__global__ void prep_step_kernel(const float* __restrict__ bih, const float* __restrict__ bhh,
                                 const float* __restrict__ Whh, const float* __restrict__ sg,
                                 float* __restrict__ biasvec, float* __restrict__ svec) {
  int j = blockIdx.x * blockDim.x + threadIdx.x;
  if (j < G4) {
    biasvec[j] = bih[j] + bhh[j];
    float s = 0.f;
    for (int k = 0; k < DM2; ++k) s += sg[k] * Whh[(long)j * HH + DM2 + k];
    svec[j] = s;
  }
}

// --------------- WMMA GEMM: C = A(M,256) @ B(N,256)^T + bias + add ---------
__global__ __launch_bounds__(256)
void wmma_gemm_kernel(const _Float16* __restrict__ A,   // (M, 256) f16
                      const _Float16* __restrict__ Bm,  // (N, 256) f16 [n][k]
                      const float* __restrict__ colBias,// (N) or null
                      const float* __restrict__ addMat, // (M, N) or null
                      float* __restrict__ C, int N) {
  __shared__ __align__(16) _Float16 As[64][264];
  const int t = threadIdx.x;
  const long rb = (long)blockIdx.x * 64;
  const int cb = blockIdx.y * 128;
  {
    int row = t >> 2, part = t & 3;
    const v8h* src = (const v8h*)(A + (rb + row) * DM2 + part * 64);
    v8h* dst = (v8h*)&As[row][part * 64];
#pragma unroll
    for (int i = 0; i < 8; ++i) dst[i] = src[i];
  }
  __syncthreads();

  const int wave = t >> 5, lane = t & 31;
  const int nlo = lane & 15;
  const bool hiHalf = (lane & 16) != 0;
  const int col = cb + wave * 16 + nlo;

  v16h bf[8];
  {
    const _Float16* base = Bm + (long)col * DM2 + (hiHalf ? 16 : 0);
#pragma unroll
    for (int kb = 0; kb < 8; ++kb) {
      const _Float16* p = base + kb * 32;
      V16U u; u.h[0] = *(const v8h*)p; u.h[1] = *(const v8h*)(p + 8);
      bf[kb] = u.v;
    }
  }
  const float cBias = colBias ? colBias[col] : 0.f;

#pragma unroll
  for (int mt = 0; mt < 4; ++mt) {
    v8f acc = {};
#pragma unroll
    for (int kb = 0; kb < 8; ++kb) {
      const _Float16* p = &As[mt * 16 + nlo][kb * 32 + (hiHalf ? 16 : 0)];
      V16U u; u.h[0] = *(const v8h*)p; u.h[1] = *(const v8h*)(p + 8);
      acc = __builtin_amdgcn_wmma_f32_16x16x32_f16(false, u.v, false, bf[kb],
                                                   (short)0, acc, false, false);
    }
#pragma unroll
    for (int r = 0; r < 8; ++r) {
      long rowm = rb + mt * 16 + r + (hiHalf ? 8 : 0);
      float v = acc[r] + cBias;
      if (addMat) v += addMat[rowm * N + col];
      C[rowm * N + col] = v;
    }
  }
}

// ------------------------------- LSTM cell ---------------------------------
__global__ void lstm_cell_kernel(const float* __restrict__ g, float* __restrict__ c,
                                 const float* __restrict__ qv,
                                 float* __restrict__ hq, _Float16* __restrict__ hqh,
                                 int first, long total) {
  long i = (long)blockIdx.x * blockDim.x + threadIdx.x;
  long st = (long)gridDim.x * blockDim.x;
  for (; i < total; i += st) {
    long b = i >> 9; int j = (int)(i & 511);
    const float* gb = g + b * G4;
    float ig = gb[j], fg = gb[HH + j], gg = gb[2 * HH + j], og = gb[3 * HH + j];
    float cold = first ? 0.f : c[i];
    float cn = sigf(fg) * cold + sigf(ig) * tanhf(gg);
    c[i] = cn;
    float hh = sigf(og) * tanhf(cn);
    if (j < DM2) {
      float v = qv[b * DM2 + j] + hh;
      hq[b * DM2 + j] = v;
      hqh[b * DM2 + j] = (_Float16)v;
    }
  }
}

// -------------------------- out[b] = hq[b] . sg ----------------------------
__global__ __launch_bounds__(256)
void final_dot_kernel(const float* __restrict__ hq, const float* __restrict__ sg,
                      float* __restrict__ out) {
  __shared__ float red[256];
  int b = blockIdx.x, t = threadIdx.x;
  red[t] = hq[(long)b * DM2 + t] * sg[t];
  __syncthreads();
  for (int s = 128; s > 0; s >>= 1) {
    if (t < s) red[t] += red[t + s];
    __syncthreads();
  }
  if (t == 0) out[b] = red[0];
}

// ---------------------------------------------------------------------------
extern "C" void kernel_launch(void* const* d_in, const int* in_sizes, int n_in,
                              void* d_out, int out_size, void* d_ws, size_t ws_size,
                              hipStream_t stream) {
  (void)in_sizes; (void)n_in; (void)out_size; (void)ws_size;
  const int*   query   = (const int*)d_in[0];
  const int*   support = (const int*)d_in[1];
  const int*   q_l1 = (const int*)d_in[2];
  const int*   q_l2 = (const int*)d_in[3];
  const int*   q_r1 = (const int*)d_in[5];
  const int*   q_r2 = (const int*)d_in[6];
  const int*   s_l1 = (const int*)d_in[8];
  const int*   s_l2 = (const int*)d_in[9];
  const int*   s_r1 = (const int*)d_in[11];
  const int*   s_r2 = (const int*)d_in[12];
  const float* emb    = (const float*)d_in[14];
  const float* gcn_W  = (const float*)d_in[15];
  const float* gcn_lb = (const float*)d_in[16];
  const float* gcn_b  = (const float*)d_in[17];
  const float* attn_W = (const float*)d_in[18];
  const float* attn_b = (const float*)d_in[19];
  const float* gate_W = (const float*)d_in[20];
  const float* gate_lb= (const float*)d_in[21];
  const float* gate_b = (const float*)d_in[22];
  const float* p1_W = (const float*)d_in[23];
  const float* p1_b = (const float*)d_in[24];
  const float* p2_W = (const float*)d_in[25];
  const float* p2_b = (const float*)d_in[26];
  const float* ln_a = (const float*)d_in[27];
  const float* ln_b = (const float*)d_in[28];
  const float* Wih  = (const float*)d_in[29];
  const float* Whh  = (const float*)d_in[30];
  const float* bih  = (const float*)d_in[31];
  const float* bhh  = (const float*)d_in[32];
  float* out = (float*)d_out;

  // ---- workspace layout (~150 MB) ----
  char* ws = (char*)d_ws;
  auto alloc = [&](size_t bytes) -> char* {
    char* p = ws; ws += (bytes + 255) & ~(size_t)255; return p;
  };
  _Float16* emb_h   = (_Float16*)alloc((size_t)(NV + 1) * DD * 2);  // 51.2 MB
  _Float16* gcnW_h  = (_Float16*)alloc((size_t)DD * DM2 * 2);
  _Float16* Wih_h   = (_Float16*)alloc((size_t)G4 * DM2 * 2);
  _Float16* WhhL_h  = (_Float16*)alloc((size_t)G4 * DM2 * 2);
  float* qn[4]; for (int i = 0; i < 4; ++i) qn[i] = (float*)alloc((size_t)BQ * DD * 4);
  float* sn[4]; for (int i = 0; i < 4; ++i) sn[i] = (float*)alloc((size_t)DD * 4);
  float*    qv     = (float*)alloc((size_t)BQ * DM2 * 4);
  _Float16* qv_h   = (_Float16*)alloc((size_t)BQ * DM2 * 2);
  float*    sv     = (float*)alloc((size_t)DM2 * 4);
  _Float16* sv_h   = (_Float16*)alloc((size_t)DM2 * 2);   // unused scratch
  float*    sg     = (float*)alloc((size_t)DM2 * 4);
  float*    biasv  = (float*)alloc((size_t)G4 * 4);
  float*    svec   = (float*)alloc((size_t)G4 * 4);
  float*    gbase0 = (float*)alloc((size_t)BQ * G4 * 4);  // 32 MB
  float*    gbuf   = (float*)alloc((size_t)BQ * G4 * 4);  // 32 MB
  float*    cbuf   = (float*)alloc((size_t)BQ * HH * 4);
  float*    hq     = (float*)alloc((size_t)BQ * DM2 * 4);
  _Float16* hq_h   = (_Float16*)alloc((size_t)BQ * DM2 * 2);

  // ---- 1. precision conversion (emb table -> f16, L2 resident) ----
  f32_to_f16_kernel<<<4096, 256, 0, stream>>>(emb, emb_h, (long)(NV + 1) * DD);
  f32_to_f16_kernel<<<128, 256, 0, stream>>>(gcn_W, gcnW_h, (long)DD * DM2);
  f32_to_f16_kernel<<<2048, 256, 0, stream>>>(Wih, Wih_h, (long)G4 * DM2);
  f32_to_f16_strided_kernel<<<2048, 256, 0, stream>>>(Whh, WhhL_h, G4, DM2, HH);

  // ---- 2. fused nbr (gather + WMMA GEMM + attention) ----
  nbr_wmma_kernel<<<BQ, 256, 0, stream>>>(q_l1, query, 0, emb_h, emb, gcnW_h,
      gcn_lb, gcn_b, attn_W, attn_b, gate_W, gate_lb, gate_b, qn[0]);
  nbr_wmma_kernel<<<BQ, 256, 0, stream>>>(q_l2, query, 0, emb_h, emb, gcnW_h,
      gcn_lb, gcn_b, attn_W, attn_b, gate_W, gate_lb, gate_b, qn[1]);
  nbr_wmma_kernel<<<BQ, 256, 0, stream>>>(q_r1, query, 1, emb_h, emb, gcnW_h,
      gcn_lb, gcn_b, attn_W, attn_b, gate_W, gate_lb, gate_b, qn[2]);
  nbr_wmma_kernel<<<BQ, 256, 0, stream>>>(q_r2, query, 1, emb_h, emb, gcnW_h,
      gcn_lb, gcn_b, attn_W, attn_b, gate_W, gate_lb, gate_b, qn[3]);
  nbr_wmma_kernel<<<1, 256, 0, stream>>>(s_l1, support, 0, emb_h, emb, gcnW_h,
      gcn_lb, gcn_b, attn_W, attn_b, gate_W, gate_lb, gate_b, sn[0]);
  nbr_wmma_kernel<<<1, 256, 0, stream>>>(s_l2, support, 0, emb_h, emb, gcnW_h,
      gcn_lb, gcn_b, attn_W, attn_b, gate_W, gate_lb, gate_b, sn[1]);
  nbr_wmma_kernel<<<1, 256, 0, stream>>>(s_r1, support, 1, emb_h, emb, gcnW_h,
      gcn_lb, gcn_b, attn_W, attn_b, gate_W, gate_lb, gate_b, sn[2]);
  nbr_wmma_kernel<<<1, 256, 0, stream>>>(s_r2, support, 1, emb_h, emb, gcnW_h,
      gcn_lb, gcn_b, attn_W, attn_b, gate_W, gate_lb, gate_b, sn[3]);

  // ---- 3. pair vectors ----
  combine_pair_kernel<<<2048, 256, 0, stream>>>(qn[0], qn[1], qn[2], qn[3], qv, qv_h, (long)BQ);
  combine_pair_kernel<<<1, 256, 0, stream>>>(sn[0], sn[1], sn[2], sn[3], sv, sv_h, (long)1);

  // ---- 4. support MLP + LayerNorm -> sg; constant step vectors ----
  support_mlp_kernel<<<1, 512, 0, stream>>>(sv, p1_W, p1_b, p2_W, p2_b, ln_a, ln_b, sg);
  prep_step_kernel<<<G4 / 256, 256, 0, stream>>>(bih, bhh, Whh, sg, biasv, svec);

  // ---- 5. gbase0 = qv @ Wih.T + (bih + bhh)  (step-invariant) ----
  {
    dim3 grid(BQ / 64, G4 / 128);
    wmma_gemm_kernel<<<grid, 256, 0, stream>>>(qv_h, Wih_h, biasv, nullptr, gbase0, G4);
  }

  // ---- 6. recurrence: step 1 uses gbase0 directly (h_r == 0) ----
  const long lstmTotal = (long)BQ * HH;
  lstm_cell_kernel<<<2048, 256, 0, stream>>>(gbase0, cbuf, qv, hq, hq_h, 1, lstmTotal);
  for (int s = 1; s < 4; ++s) {
    dim3 grid(BQ / 64, G4 / 128);
    // g = gbase0 + svec + hq @ WhhL.T   (svec = sg @ WhhR.T, constant)
    wmma_gemm_kernel<<<grid, 256, 0, stream>>>(hq_h, WhhL_h, svec, gbase0, gbuf, G4);
    lstm_cell_kernel<<<2048, 256, 0, stream>>>(gbuf, cbuf, qv, hq, hq_h, 0, lstmTotal);
  }

  // ---- 7. out[b] = sum(hq[b] * sg) ----
  final_dot_kernel<<<BQ, 256, 0, stream>>>(hq, sg, out);
}
